// AttentionLayer_4690104287306
// MI455X (gfx1250) — matrix operand order repaired and checked by reading, not compiled, not run on hardware
//
#include <hip/hip_runtime.h>

typedef __attribute__((ext_vector_type(16))) __bf16 v16bf;
typedef __attribute__((ext_vector_type(8)))  __bf16 v8bf;
typedef __attribute__((ext_vector_type(8)))  float  v8f;

#define BATCH  2
#define SEQ    2048
#define DMODEL 1024
#define NHEAD  16
#define HDIM   64
#define MTOT   (BATCH * SEQ)

__device__ __forceinline__ __bf16 f2bf(float f) { return (__bf16)f; }

__device__ __forceinline__ v16bf pack16(v8bf lo, v8bf hi) {
  v16bf o;
#pragma unroll
  for (int i = 0; i < 8; ++i) { o[i] = lo[i]; o[8 + i] = hi[i]; }
  return o;
}

// --- gfx1250 async global->LDS copy (ASYNCcnt path), with sync fallback ----
#if defined(__gfx1250__) && __has_builtin(__builtin_amdgcn_global_load_async_to_lds_b128)
#define HAVE_ASYNC_LDS 1
typedef int v4i_ __attribute__((vector_size(16)));          // GCC-style v4i
typedef __attribute__((address_space(1))) v4i_* gas1_v4i;   // global (AS1)
typedef __attribute__((address_space(3))) v4i_* las3_v4i;   // LDS (AS3)
__device__ __forceinline__ void async_copy_b128(void* lds_dst, const void* gsrc) {
  // generic->AS1 via integer round-trip (global generic == AS1 numerically);
  // low 32 bits of a generic LDS address are the LDS byte offset (AS3).
  gas1_v4i g = (gas1_v4i)(uintptr_t)gsrc;
  las3_v4i l = (las3_v4i)(unsigned)(uintptr_t)lds_dst;
  __builtin_amdgcn_global_load_async_to_lds_b128(g, l, 0, 0);
}
__device__ __forceinline__ void wait_async() {
#if __has_builtin(__builtin_amdgcn_s_wait_asynccnt)
  __builtin_amdgcn_s_wait_asynccnt(0);
#else
  asm volatile("s_wait_asynccnt 0x0" ::: "memory");
#endif
}
#else
#define HAVE_ASYNC_LDS 0
__device__ __forceinline__ void async_copy_b128(void* lds_dst, const void* gsrc) {
  *(v8bf*)lds_dst = *(const v8bf*)gsrc;
}
__device__ __forceinline__ void wait_async() {}
#endif

// ---------------------------------------------------------------------------
// Pass 0a: W[D][D] f32 -> Wt[n][k] bf16 (transposed, n-major)
// ---------------------------------------------------------------------------
__global__ void __launch_bounds__(256)
prep_w_kernel(const float* __restrict__ Wq, const float* __restrict__ Wk,
              const float* __restrict__ Wv, __bf16* __restrict__ Wt) {
  const int e = blockIdx.x * 256 + threadIdx.x;   // 0 .. D*D-1
  const int z = blockIdx.y;
  const float* W = (z == 0) ? Wq : (z == 1) ? Wk : Wv;
  const int i = e >> 10;                          // k
  const int j = e & (DMODEL - 1);                 // n
  Wt[(size_t)z * DMODEL * DMODEL + (size_t)j * DMODEL + i] =
      f2bf(W[(size_t)i * DMODEL + j]);
}

// ---------------------------------------------------------------------------
// Pass 0b: q/k/v f32 -> Xb bf16 (same layout)
// ---------------------------------------------------------------------------
__global__ void __launch_bounds__(256)
prep_x_kernel(const float* __restrict__ q, const float* __restrict__ k,
              const float* __restrict__ v, __bf16* __restrict__ Xb) {
  const size_t e = (size_t)blockIdx.x * 256 + threadIdx.x;  // 0 .. MTOT*D-1
  const int z = blockIdx.y;
  const float* x = (z == 0) ? q : (z == 1) ? k : v;
  Xb[(size_t)z * MTOT * DMODEL + e] = f2bf(x[e]);
}

// ---------------------------------------------------------------------------
// Pass 1: Y = X @ W + b, split by head.  grid = (N/64, M/256, 3)
// Each wave: 32x64 tile (2 A-frags, 4 shared B-frags, 8 WMMAs per k-step).
//   z=0 -> Qh[B,H,S,HD]  z=1 -> Kh[B,H,S,HD]  z=2 -> Vt[B,H,HD,S]
// ---------------------------------------------------------------------------
__global__ void __launch_bounds__(256)
proj_qkv_kernel(const __bf16* __restrict__ Xb,
                const float* __restrict__ bq, const float* __restrict__ bk,
                const float* __restrict__ bv,
                const __bf16* __restrict__ Wt,
                __bf16* __restrict__ Qh, __bf16* __restrict__ Kh,
                __bf16* __restrict__ Vt) {
  const int lane = threadIdx.x & 31;
  const int wave = threadIdx.x >> 5;
  const int z    = blockIdx.z;

  const __bf16* x   = Xb + (size_t)z * MTOT * DMODEL;
  const float* bias = (z == 0) ? bq : (z == 1) ? bk : bv;
  const __bf16* W   = Wt + (size_t)z * DMODEL * DMODEL;

  const int mbase = (blockIdx.y * 8 + wave) * 32;
  const int nbase = blockIdx.x * 64;
  const int col   = lane & 15;
  const int ka    = (lane < 16) ? 0 : 16;   // A-fragment K base
  const int kb    = (lane < 16) ? 0 : 16;   // B-fragment K base
  const int rofs  = (lane >> 4) << 3;       // C-matrix row offset (0 or 8)

  const __bf16* xrow0 = x + (size_t)(mbase + col) * DMODEL;
  const __bf16* xrow1 = x + (size_t)(mbase + 16 + col) * DMODEL;

  v8f acc[2][4] = {};

  for (int kk = 0; kk < DMODEL; kk += 32) {
    v16bf a0 = pack16(*(const v8bf*)(xrow0 + kk + ka),
                      *(const v8bf*)(xrow0 + kk + ka + 8));
    v16bf a1 = pack16(*(const v8bf*)(xrow1 + kk + ka),
                      *(const v8bf*)(xrow1 + kk + ka + 8));
#pragma unroll
    for (int nt = 0; nt < 4; ++nt) {
      v16bf bfrag =
          *(const v16bf*)(W + (size_t)(nbase + nt * 16 + col) * DMODEL + kk + kb);
      acc[0][nt] = __builtin_amdgcn_wmma_f32_16x16x32_bf16(
          false, a0, false, bfrag, (short)0, acc[0][nt], false, false);
      acc[1][nt] = __builtin_amdgcn_wmma_f32_16x16x32_bf16(
          false, a1, false, bfrag, (short)0, acc[1][nt], false, false);
    }
  }

#pragma unroll
  for (int nt = 0; nt < 4; ++nt) {
    const int n  = nbase + nt * 16 + col;
    const float bn = bias[n];
    const int h  = n >> 6;
    const int hd = n & (HDIM - 1);
#pragma unroll
    for (int mt = 0; mt < 2; ++mt) {
#pragma unroll
      for (int r = 0; r < 8; ++r) {
        const int m  = mbase + mt * 16 + r + rofs;
        const int bb = m >> 11;
        const int s  = m & (SEQ - 1);
        const __bf16 val = f2bf(acc[mt][nt][r] + bn);
        if (z == 2) {
          Vt[(((size_t)bb * NHEAD + h) * HDIM + hd) * SEQ + s] = val;
        } else {
          __bf16* dst = (z == 0) ? Qh : Kh;
          dst[(((size_t)bb * NHEAD + h) * SEQ + s) * HDIM + hd] = val;
        }
      }
    }
  }
}

// ---------------------------------------------------------------------------
// Pass 2: flash attention.  Block = 8 waves sharing one (b,h); each wave owns
// a 16-query tile; K/V chunks (32 keys) staged cooperatively in LDS via
// async global->LDS copies; next chunk prefetched during compute.
// ---------------------------------------------------------------------------
__global__ void __launch_bounds__(256)
attn_kernel(const __bf16* __restrict__ Qh, const __bf16* __restrict__ Kh,
            const __bf16* __restrict__ Vt, float* __restrict__ out) {
  __shared__ __attribute__((aligned(64))) __bf16 Kbuf[32 * 64];   // [key][hd]
  __shared__ __attribute__((aligned(64))) __bf16 Vbuf[64 * 32];   // [hd][key]
  __shared__ __attribute__((aligned(64))) __bf16 ldsP[8][16 * 32];

  const int lane = threadIdx.x & 31;
  const int wave = threadIdx.x >> 5;
  const int qblk = blockIdx.x & 15;          // 16 q-blocks of 128
  const int h    = (blockIdx.x >> 4) & 15;
  const int b    = blockIdx.x >> 8;

  const int qbase = qblk * 128 + wave * 16;
  const int col   = lane & 15;
  const int ka    = (lane < 16) ? 0 : 16;
  const int kb    = (lane < 16) ? 0 : 16;
  const int rofs  = (lane >> 4) << 3;

  const __bf16* kbp = Kh + ((size_t)b * NHEAD + h) * SEQ * HDIM;
  const __bf16* vbp = Vt + ((size_t)b * NHEAD + h) * HDIM * SEQ;

  // staging assignments: 256 threads move 8 KB (K: 32x64, V: 64x32)
  const __bf16* ksrc = kbp + (size_t)(threadIdx.x >> 3) * HDIM + (threadIdx.x & 7) * 8;
  __bf16*       kdst = Kbuf + (threadIdx.x >> 3) * 64 + (threadIdx.x & 7) * 8;
  const __bf16* vsrc = vbp + (size_t)(threadIdx.x >> 2) * SEQ + (threadIdx.x & 3) * 8;
  __bf16*       vdst = Vbuf + (threadIdx.x >> 2) * 32 + (threadIdx.x & 3) * 8;

  // resident Q fragments (K-dim 0..31 / 32..63)
  const __bf16* qrow = Qh + (((size_t)b * NHEAD + h) * SEQ + qbase + col) * HDIM;
  v16bf aQ0 = pack16(*(const v8bf*)(qrow + ka), *(const v8bf*)(qrow + ka + 8));
  v16bf aQ1 = pack16(*(const v8bf*)(qrow + 32 + ka),
                     *(const v8bf*)(qrow + 32 + ka + 8));

  float mrow[8], lrow[8];
#pragma unroll
  for (int r = 0; r < 8; ++r) { mrow[r] = -1e30f; lrow[r] = 0.f; }
  v8f o[4] = {};

  for (int kc = 0; kc < SEQ; kc += 32) {
    // ---- cooperative staging of K/V chunk (ASYNCcnt path) ----
    __syncthreads();
    async_copy_b128(kdst, ksrc + (size_t)kc * HDIM);
    async_copy_b128(vdst, vsrc + kc);
    wait_async();
    __syncthreads();

    if (kc + 32 < SEQ) {   // uniform branch: prefetch next chunk into caches
      __builtin_prefetch(ksrc + (size_t)(kc + 32) * HDIM, 0, 0);
      __builtin_prefetch(vsrc + kc + 32, 0, 0);
    }

    // ---- scores: 16 queries x 32 keys ----
    v8f st[2];
#pragma unroll
    for (int t = 0; t < 2; ++t) {
      const __bf16* krow = Kbuf + (t * 16 + col) * 64;
      v16bf bk0 = *(const v16bf*)(krow + kb);
      v16bf bk1 = *(const v16bf*)(krow + 32 + kb);
      v8f s = {};
      s = __builtin_amdgcn_wmma_f32_16x16x32_bf16(false, aQ0, false, bk0,
                                                  (short)0, s, false, false);
      s = __builtin_amdgcn_wmma_f32_16x16x32_bf16(false, aQ1, false, bk1,
                                                  (short)0, s, false, false);
      st[t] = s * 0.125f;                 // 1/sqrt(HD)
    }

    // ---- online softmax ----
    float mnew[8], alpha[8], p0[8], p1[8];
#pragma unroll
    for (int r = 0; r < 8; ++r) {
      float mx = fmaxf(st[0][r], st[1][r]);
#pragma unroll
      for (int d = 1; d < 16; d <<= 1) mx = fmaxf(mx, __shfl_xor(mx, d, 32));
      mnew[r]  = fmaxf(mrow[r], mx);
      alpha[r] = __expf(mrow[r] - mnew[r]);
    }
#pragma unroll
    for (int r = 0; r < 8; ++r) {
      p0[r] = __expf(st[0][r] - mnew[r]);
      p1[r] = __expf(st[1][r] - mnew[r]);
      float rs = p0[r] + p1[r];
#pragma unroll
      for (int d = 1; d < 16; d <<= 1) rs += __shfl_xor(rs, d, 32);
      lrow[r] = lrow[r] * alpha[r] + rs;
      mrow[r] = mnew[r];
    }
#pragma unroll
    for (int nt = 0; nt < 4; ++nt)
#pragma unroll
      for (int r = 0; r < 8; ++r) o[nt][r] *= alpha[r];

    // ---- transpose P (C layout -> A layout) via per-wave LDS ----
    __bf16* lp = ldsP[wave];
#pragma unroll
    for (int r = 0; r < 8; ++r) {
      lp[(r + rofs) * 32 + col]      = f2bf(p0[r]);
      lp[(r + rofs) * 32 + 16 + col] = f2bf(p1[r]);
    }
    const __bf16* pr = lp + col * 32 + ka;
    v16bf aP = pack16(*(const v8bf*)(pr), *(const v8bf*)(pr + 8));

    // ---- O += P @ V ----
#pragma unroll
    for (int nt = 0; nt < 4; ++nt) {
      v16bf bV = *(const v16bf*)(Vbuf + (nt * 16 + col) * 32 + kb);
      o[nt] = __builtin_amdgcn_wmma_f32_16x16x32_bf16(
          false, aP, false, bV, (short)0, o[nt], false, false);
    }
  }

  // ---- epilogue ----
  float inv[8];
#pragma unroll
  for (int r = 0; r < 8; ++r) inv[r] = 1.0f / lrow[r];
#pragma unroll
  for (int nt = 0; nt < 4; ++nt) {
    const int dim = nt * 16 + col;
#pragma unroll
    for (int r = 0; r < 8; ++r) {
      const int s = qbase + r + rofs;
      out[((size_t)b * SEQ + s) * DMODEL + h * HDIM + dim] = o[nt][r] * inv[r];
    }
  }
}

// ---------------------------------------------------------------------------
extern "C" void kernel_launch(void* const* d_in, const int* in_sizes, int n_in,
                              void* d_out, int out_size, void* d_ws,
                              size_t ws_size, hipStream_t stream) {
  (void)in_sizes; (void)n_in; (void)out_size; (void)ws_size;
  const float* q  = (const float*)d_in[0];
  const float* k  = (const float*)d_in[1];
  const float* v  = (const float*)d_in[2];
  const float* Wq = (const float*)d_in[3];
  const float* bq = (const float*)d_in[4];
  const float* Wk = (const float*)d_in[5];
  const float* bk = (const float*)d_in[6];
  const float* Wv = (const float*)d_in[7];
  const float* bv = (const float*)d_in[8];
  float* out = (float*)d_out;

  // ws layout (bf16): Wt[3][D][D] | Xb[3][M][D] | Qh | Kh | Vt  (~54 MB)
  char* ws = (char*)d_ws;
  __bf16* Wt = (__bf16*)ws;
  __bf16* Xb = Wt + (size_t)3 * DMODEL * DMODEL;
  __bf16* Qh = Xb + (size_t)3 * MTOT * DMODEL;
  __bf16* Kh = Qh + (size_t)MTOT * DMODEL;
  __bf16* Vt = Kh + (size_t)MTOT * DMODEL;

  prep_w_kernel<<<dim3(DMODEL * DMODEL / 256, 3), 256, 0, stream>>>(Wq, Wk, Wv, Wt);
  prep_x_kernel<<<dim3(MTOT * DMODEL / 256, 3), 256, 0, stream>>>(q, k, v, Xb);
  proj_qkv_kernel<<<dim3(DMODEL / 64, MTOT / 256, 3), 256, 0, stream>>>(
      Xb, bq, bk, bv, Wt, Qh, Kh, Vt);
  attn_kernel<<<dim3(BATCH * NHEAD * (SEQ / 128)), 256, 0, stream>>>(
      Qh, Kh, Vt, out);
}
